// MyDCGRU_73065983640239
// MI455X (gfx1250) — compile-verified, matrix operand order checked by reading, not run on hardware
//
#include <hip/hip_runtime.h>
#include <math.h>

// ---------------------------------------------------------------------------
// DCGRU encoder forward for MI455X (gfx1250), bf16 WMMA + f32 accumulate.
//
// Dominant work: 72 dense GEMMs of [2048x2048]x[2048x2048] (support diffusion;
// the batched einsum collapses into one GEMM because the support matrix is
// shared across the batch). All operands are L2-resident (8-16 MB each vs
// 192 MB L2) -> compute bound -> v_wmma_f32_16x16x32_bf16 (8x per-instruction
// FLOPs of the f32 WMMA; fp8 rejected for 12-step recurrent precision).
//
// Diffusion GEMM: 128x128 block tile, 4 waves in a 2x2 grid, 64x64 wave tile
// (4x4 16x16 fragments -> 16 WMMA per 16 ds_load_b128 per k-tile), LDS
// double-buffered (one barrier per k-tile), padded LDS stride 40 bf16 for a
// conflict-minimal b128 access pattern.
// ---------------------------------------------------------------------------

typedef __attribute__((ext_vector_type(8)))  float   v8f;
typedef __attribute__((ext_vector_type(8)))  __bf16  v8bf;
typedef __attribute__((ext_vector_type(16))) __bf16  v16bf;

#define N_NODES 2048
#define BATCH   16
#define NP      12
#define NQ      12
#define ND      64
#define F2      128          // 2*D (concat width)
#define NCOLC   2048         // BATCH * F2 (GEMM column count)
#define NROWS   32768        // N_NODES * BATCH
#define LDSTR   40           // padded LDS row stride in bf16 (32 data + 8 pad)
#define KTILES  (N_NODES / 32)

static __device__ __forceinline__ v8f vzero8() {
  v8f v;
#pragma unroll
  for (int i = 0; i < 8; ++i) v[i] = 0.0f;
  return v;
}
static __device__ __forceinline__ v16bf vcat(v8bf lo, v8bf hi) {
  return __builtin_shufflevector(lo, hi, 0,1,2,3,4,5,6,7,8,9,10,11,12,13,14,15);
}
static __device__ __forceinline__ v8f wmma_bf16(v16bf a, v16bf b, v8f c) {
  // (neg_a, A, neg_b, B, c_mod, C, reuse_a, reuse_b)
  return __builtin_amdgcn_wmma_f32_16x16x32_bf16(false, a, false, b, (short)0, c,
                                                 false, false);
}

// --------------------------- setup kernels ---------------------------------

__global__ __launch_bounds__(256) void k_cast_support(const float* __restrict__ S,
                                                      __bf16* __restrict__ Sb) {
  int i = blockIdx.x * 256 + threadIdx.x;        // N_NODES*N_NODES exact
  Sb[i] = (__bf16)S[i];
}

__global__ __launch_bounds__(256) void k_prep_w(const float* __restrict__ Wg,
                                                const float* __restrict__ Wc,
                                                __bf16* __restrict__ WgT,
                                                __bf16* __restrict__ WcT) {
  int i = blockIdx.x * 256 + threadIdx.x;        // 512*128 + 512*64 exact
  if (i < 512 * 128) {
    int k = i / 128, j = i % 128;
    WgT[(size_t)j * 512 + k] = (__bf16)Wg[i];
  } else {
    int i2 = i - 512 * 128;
    int k = i2 / 64, j = i2 % 64;
    WcT[(size_t)j * 512 + k] = (__bf16)Wc[i2];
  }
}

__global__ __launch_bounds__(256) void k_zero(float* __restrict__ p) {
  p[blockIdx.x * 256 + threadIdx.x] = 0.0f;
}

// Per-step input embedding: xt[(n,b)][d] = (relu(x*W1+b1) @ W2 + b2), bf16.
__global__ __launch_bounds__(64) void k_xemb(const float* __restrict__ X,
                                             const float* __restrict__ W1,
                                             const float* __restrict__ b1,
                                             const float* __restrict__ W2,
                                             const float* __restrict__ b2,
                                             int t, __bf16* __restrict__ xt) {
  __shared__ float hs[ND];
  const int nb = blockIdx.x;                 // n*16 + b
  const int d  = threadIdx.x;
  const int n = nb >> 4, b = nb & 15;
  const float x = X[((size_t)b * NP + t) * N_NODES + n];
  hs[d] = fmaxf(x * W1[d] + b1[d], 0.0f);
  __syncthreads();
  float acc = b2[d];
#pragma unroll 8
  for (int k = 0; k < ND; ++k) acc += hs[k] * W2[k * ND + d];
  xt[(size_t)nb * ND + d] = (__bf16)acc;
}

// Build Z0 = concat([xt, h]) in both layouts.
__global__ __launch_bounds__(256) void k_build_z0(const __bf16* __restrict__ xt,
                                                  const float* __restrict__ H,
                                                  __bf16* __restrict__ Z0,
                                                  __bf16* __restrict__ ZT0) {
  int i = blockIdx.x * 256 + threadIdx.x;        // N_NODES*NCOLC exact
  int n = i >> 11, c = i & 2047;
  int b = c >> 7, f = c & 127;
  int nb = (n << 4) | b;
  __bf16 v = (f < ND) ? xt[(size_t)nb * ND + f]
                      : (__bf16)H[(size_t)nb * ND + (f - ND)];
  Z0 [(size_t)n * NCOLC + c] = v;
  ZT0[(size_t)c * N_NODES + n] = v;
}

// Overwrite state half of C0 with r*h (xt half is still in place from k_build_z0).
__global__ __launch_bounds__(256) void k_gate_apply(const float* __restrict__ RU,
                                                    const float* __restrict__ H,
                                                    __bf16* __restrict__ Z0,
                                                    __bf16* __restrict__ ZT0) {
  int i = blockIdx.x * 256 + threadIdx.x;        // NROWS*ND exact
  int nb = i >> 6, f = i & 63;
  float r = RU[(size_t)nb * F2 + f];             // already sigmoided
  __bf16 v = (__bf16)(r * H[i]);
  int n = nb >> 4, b = nb & 15;
  int c = (b << 7) + ND + f;
  Z0 [(size_t)n * NCOLC + c] = v;
  ZT0[(size_t)c * N_NODES + n] = v;
}

// h = u*h + (1-u)*c
__global__ __launch_bounds__(256) void k_h_update(const float* __restrict__ RU,
                                                  const float* __restrict__ Craw,
                                                  float* __restrict__ H) {
  int i = blockIdx.x * 256 + threadIdx.x;        // NROWS*ND exact
  int nb = i >> 6, f = i & 63;
  float u = RU[(size_t)nb * F2 + ND + f];
  float c = Craw[i];                              // tanh already applied
  H[i] = u * H[i] + (1.0f - u) * c;
}

// Output head: y = relu(h @ Wo1 + bo1) @ Wo2 + bo2, written as [B,Q,N].
__global__ __launch_bounds__(64) void k_head(const float* __restrict__ H,
                                             const float* __restrict__ Wo1,
                                             const float* __restrict__ bo1,
                                             const float* __restrict__ Wo2,
                                             const float* __restrict__ bo2,
                                             float* __restrict__ out) {
  __shared__ float g[ND];
  const int nb = blockIdx.x;
  const int d  = threadIdx.x;
  const int n = nb >> 4, b = nb & 15;
  const float* h = H + (size_t)nb * ND;
  float acc = bo1[d];
#pragma unroll 8
  for (int k = 0; k < ND; ++k) acc += h[k] * Wo1[k * ND + d];
  g[d] = fmaxf(acc, 0.0f);
  __syncthreads();
  if (d < NQ) {
    float y = bo2[d];
#pragma unroll 8
    for (int k = 0; k < ND; ++k) y += g[k] * Wo2[k * NQ + d];
    out[((size_t)b * NQ + d) * N_NODES + n] = y;
  }
}

// ------------------------- diffusion GEMM (WMMA) ----------------------------
// Out[n][c] = alpha * sum_m S[n][m] * Z_in[m][c] + beta * Z_prev[n][c]
// Operands: S row-major bf16 (A), ZTin[c][m] bf16 (B via transposed copy).
// 128 threads = 4 waves in a 2x2 grid, each wave owns a 64x64 output tile
// (4x4 WMMA fragments). Double-buffered LDS, one barrier per 32-wide k-tile.
__global__ __launch_bounds__(128) void k_diffuse(
    const __bf16* __restrict__ S, const __bf16* __restrict__ ZTin,
    const __bf16* __restrict__ ZTprev, float alpha, float beta,
    __bf16* __restrict__ Zout, __bf16* __restrict__ ZTout) {
  __shared__ __bf16 ldsA[2][128 * LDSTR];
  __shared__ __bf16 ldsB[2][128 * LDSTR];

  const int tid  = threadIdx.x;
  const int lane = tid & 31;
  const int wave = tid >> 5;
  const int wm   = wave >> 1;          // 2 row-slices of 64
  const int wc   = wave & 1;           // 2 col-slices of 64
  const int l15  = lane & 15;
  const int lh   = lane >> 4;
  const int blockN = blockIdx.x * 128; // node rows
  const int blockC = blockIdx.y * 128; // (b,f) cols

  v8f acc[4][4];
#pragma unroll
  for (int i = 0; i < 4; ++i)
#pragma unroll
    for (int j = 0; j < 4; ++j) acc[i][j] = vzero8();

  // Global staging: 512 16B chunks per 128x32 tile; 128 threads x 4 chunks.
  // Thread covers rows rq, rq+32, rq+64, rq+96 at 16B sub-chunk q.
  const int rq = tid >> 2;             // 0..31
  const int q  = tid & 3;
  const __bf16* gA = S    + (size_t)(blockN + rq) * N_NODES + q * 8;
  const __bf16* gB = ZTin + (size_t)(blockC + rq) * N_NODES + q * 8;
  const int lbase = rq * LDSTR + q * 8;

  v8bf pa[4], pb[4];
#pragma unroll
  for (int k = 0; k < 4; ++k) {
    pa[k] = *(const v8bf*)(gA + (size_t)(32 * k) * N_NODES);
    pb[k] = *(const v8bf*)(gB + (size_t)(32 * k) * N_NODES);
  }
#pragma unroll
  for (int k = 0; k < 4; ++k) {
    *(v8bf*)&ldsA[0][lbase + 32 * k * LDSTR] = pa[k];
    *(v8bf*)&ldsB[0][lbase + 32 * k * LDSTR] = pb[k];
  }
  __syncthreads();

  for (int kt = 0; kt < KTILES; ++kt) {
    const int cur = kt & 1, nxt = cur ^ 1;
    const bool more = (kt + 1) < KTILES;
    if (more) {                                   // issue next-tile loads early
      const int kb = (kt + 1) * 32;
#pragma unroll
      for (int k = 0; k < 4; ++k) {
        pa[k] = *(const v8bf*)(gA + (size_t)(32 * k) * N_NODES + kb);
        pb[k] = *(const v8bf*)(gB + (size_t)(32 * k) * N_NODES + kb);
      }
    }

    v16bf af[4];
#pragma unroll
    for (int i = 0; i < 4; ++i) {
      const int base = (wm * 64 + i * 16 + l15) * LDSTR + lh * 8;
      af[i] = vcat(*(const v8bf*)&ldsA[cur][base],
                   *(const v8bf*)&ldsA[cur][base + 16]);
    }
#pragma unroll
    for (int j = 0; j < 4; ++j) {
      const int base = (wc * 64 + j * 16 + l15) * LDSTR + lh * 8;
      v16bf bf_ = vcat(*(const v8bf*)&ldsB[cur][base],
                       *(const v8bf*)&ldsB[cur][base + 16]);
#pragma unroll
      for (int i = 0; i < 4; ++i)
        acc[i][j] = wmma_bf16(af[i], bf_, acc[i][j]);
    }

    if (more) {                                   // fill the other buffer
#pragma unroll
      for (int k = 0; k < 4; ++k) {
        *(v8bf*)&ldsA[nxt][lbase + 32 * k * LDSTR] = pa[k];
        *(v8bf*)&ldsB[nxt][lbase + 32 * k * LDSTR] = pb[k];
      }
    }
    __syncthreads();
  }

  // Epilogue: fuse Chebyshev recurrence, f32->bf16, dual-layout store.
#pragma unroll
  for (int i = 0; i < 4; ++i) {
#pragma unroll
    for (int j = 0; j < 4; ++j) {
      const int cg = blockC + wc * 64 + j * 16 + l15;       // fixed col per lane
      const int n0 = blockN + wm * 64 + i * 16 + lh * 8;    // rows n0..n0+7
      float vals[8];
#pragma unroll
      for (int v = 0; v < 8; ++v) vals[v] = alpha * acc[i][j][v];
      if (beta != 0.0f) {
        v8bf zp = *(const v8bf*)(ZTprev + (size_t)cg * N_NODES + n0);
#pragma unroll
        for (int v = 0; v < 8; ++v) vals[v] += beta * (float)zp[v];
      }
      v8bf o;
#pragma unroll
      for (int v = 0; v < 8; ++v) o[v] = (__bf16)vals[v];
      *(v8bf*)(ZTout + (size_t)cg * N_NODES + n0) = o;      // contiguous b128
#pragma unroll
      for (int v = 0; v < 8; ++v)                           // scattered b16
        Zout[(size_t)(n0 + v) * NCOLC + cg] = o[v];
    }
  }
}

// --------------------- pointwise (1x1) GEMM (WMMA) --------------------------
// out[(n,b)][j] = act( sum_{term,f} Zterm[(n,b)][f] * W[term*128+f][j] + bias[j] )
// CF = column-fragments per wave (4 -> 128 gate cols, 2 -> 64 candidate cols).
template <int CF, int ACT>
__global__ __launch_bounds__(256) void k_pointwise(
    const __bf16* __restrict__ Z0, const __bf16* __restrict__ Z1,
    const __bf16* __restrict__ Z2, const __bf16* __restrict__ Z3,
    const __bf16* __restrict__ WT, const float* __restrict__ bias,
    float* __restrict__ out) {
  constexpr int NCOLS = CF * 32;
  const int tid = threadIdx.x, lane = tid & 31, wave = tid >> 5;
  const int wm = wave >> 1, wc = wave & 1;
  const int l15 = lane & 15, lh = lane >> 4;
  const int blockM = blockIdx.x * 128;

  v8f acc[2][CF];
#pragma unroll
  for (int i = 0; i < 2; ++i)
#pragma unroll
    for (int j = 0; j < CF; ++j) acc[i][j] = vzero8();

  const __bf16* Zt[4] = {Z0, Z1, Z2, Z3};
#pragma unroll
  for (int term = 0; term < 4; ++term) {
    const __bf16* __restrict__ Zp = Zt[term];
#pragma unroll
    for (int sub = 0; sub < 4; ++sub) {
      const int f0 = sub * 32;
      v16bf afrag[2];
#pragma unroll
      for (int i = 0; i < 2; ++i) {
        const size_t base =
            (size_t)(blockM + wm * 32 + i * 16 + l15) * F2 + f0 + lh * 8;
        afrag[i] = vcat(*(const v8bf*)(Zp + base), *(const v8bf*)(Zp + base + 16));
      }
      const int kb = term * 128 + f0 + lh * 8;
#pragma unroll
      for (int j = 0; j < CF; ++j) {
        const size_t wbase = (size_t)(wc * (NCOLS / 2) + j * 16 + l15) * 512 + kb;
        v16bf bfrag = vcat(*(const v8bf*)(WT + wbase),
                           *(const v8bf*)(WT + wbase + 16));
#pragma unroll
        for (int i = 0; i < 2; ++i)
          acc[i][j] = wmma_bf16(afrag[i], bfrag, acc[i][j]);
      }
    }
  }
#pragma unroll
  for (int i = 0; i < 2; ++i) {
#pragma unroll
    for (int j = 0; j < CF; ++j) {
      const int col = wc * (NCOLS / 2) + j * 16 + l15;
      const float bv = bias[col];
      const int row0 = blockM + wm * 32 + i * 16 + lh * 8;
#pragma unroll
      for (int v = 0; v < 8; ++v) {
        float x = acc[i][j][v] + bv;
        x = (ACT == 0) ? (1.0f / (1.0f + __expf(-x))) : tanhf(x);
        out[(size_t)(row0 + v) * NCOLS + col] = x;
      }
    }
  }
}

// ------------------------------- driver -------------------------------------

extern "C" void kernel_launch(void* const* d_in, const int* in_sizes, int n_in,
                              void* d_out, int out_size, void* d_ws,
                              size_t ws_size, hipStream_t stream) {
  (void)in_sizes; (void)n_in; (void)out_size; (void)ws_size;
  const float* X   = (const float*)d_in[0];
  // d_in[1] = TE (unused by the reference forward path)
  const float* Sup = (const float*)d_in[2];
  const float* W1  = (const float*)d_in[3];
  const float* b1  = (const float*)d_in[4];
  const float* W2  = (const float*)d_in[5];
  const float* b2  = (const float*)d_in[6];
  const float* Wg  = (const float*)d_in[7];
  const float* bg  = (const float*)d_in[8];
  const float* Wc  = (const float*)d_in[9];
  const float* bc  = (const float*)d_in[10];
  const float* Wo1 = (const float*)d_in[11];
  const float* bo1 = (const float*)d_in[12];
  const float* Wo2 = (const float*)d_in[13];
  const float* bo2 = (const float*)d_in[14];
  float* out = (float*)d_out;

  // workspace layout (256B aligned)
  char* ws = (char*)d_ws;
  size_t off = 0;
  auto take = [&](size_t bytes) {
    char* p = ws + off;
    off += (bytes + 255) & ~(size_t)255;
    return p;
  };
  __bf16* Sb   = (__bf16*)take((size_t)N_NODES * N_NODES * 2);   // 8 MB
  float*  H    = (float*) take((size_t)NROWS * ND * 4);          // 8 MB
  __bf16* Z[4]; __bf16* ZT[4];
  for (int i = 0; i < 4; ++i) Z[i]  = (__bf16*)take((size_t)N_NODES * NCOLC * 2);
  for (int i = 0; i < 4; ++i) ZT[i] = (__bf16*)take((size_t)N_NODES * NCOLC * 2);
  float*  RU   = (float*) take((size_t)NROWS * F2 * 4);          // 16 MB
  float*  Craw = (float*) take((size_t)NROWS * ND * 4);          // 8 MB
  __bf16* xt   = (__bf16*)take((size_t)NROWS * ND * 2);          // 4 MB
  __bf16* WgT  = (__bf16*)take((size_t)F2 * 512 * 2);
  __bf16* WcT  = (__bf16*)take((size_t)ND * 512 * 2);

  const dim3 gGemm(16, 16);

  k_cast_support<<<(N_NODES * N_NODES) / 256, 256, 0, stream>>>(Sup, Sb);
  k_prep_w<<<(512 * 128 + 512 * 64) / 256, 256, 0, stream>>>(Wg, Wc, WgT, WcT);
  k_zero<<<(NROWS * ND) / 256, 256, 0, stream>>>(H);

  for (int t = 0; t < NP; ++t) {
    k_xemb<<<NROWS, 64, 0, stream>>>(X, W1, b1, W2, b2, t, xt);
    k_build_z0<<<(N_NODES * NCOLC) / 256, 256, 0, stream>>>(xt, H, Z[0], ZT[0]);

    // gate path diffusion: x1 = S@x0 ; x2 = 2S@x1 - x0 ; x3 = 2S@x2 - x1
    k_diffuse<<<gGemm, 128, 0, stream>>>(Sb, ZT[0], ZT[0], 1.0f,  0.0f, Z[1], ZT[1]);
    k_diffuse<<<gGemm, 128, 0, stream>>>(Sb, ZT[1], ZT[0], 2.0f, -1.0f, Z[2], ZT[2]);
    k_diffuse<<<gGemm, 128, 0, stream>>>(Sb, ZT[2], ZT[1], 2.0f, -1.0f, Z[3], ZT[3]);
    k_pointwise<4, 0><<<NROWS / 128, 256, 0, stream>>>(Z[0], Z[1], Z[2], Z[3],
                                                       WgT, bg, RU);

    // candidate path: C0 reuses Z0 buffer (xt half intact, state half = r*h)
    k_gate_apply<<<(NROWS * ND) / 256, 256, 0, stream>>>(RU, H, Z[0], ZT[0]);
    k_diffuse<<<gGemm, 128, 0, stream>>>(Sb, ZT[0], ZT[0], 1.0f,  0.0f, Z[1], ZT[1]);
    k_diffuse<<<gGemm, 128, 0, stream>>>(Sb, ZT[1], ZT[0], 2.0f, -1.0f, Z[2], ZT[2]);
    k_diffuse<<<gGemm, 128, 0, stream>>>(Sb, ZT[2], ZT[1], 2.0f, -1.0f, Z[3], ZT[3]);
    k_pointwise<2, 1><<<NROWS / 128, 256, 0, stream>>>(Z[0], Z[1], Z[2], Z[3],
                                                       WcT, bc, Craw);
    k_h_update<<<(NROWS * ND) / 256, 256, 0, stream>>>(RU, Craw, H);
  }

  k_head<<<NROWS, 64, 0, stream>>>(H, Wo1, bo1, Wo2, bo2, out);
}
